// QuantLinear_88751204204947
// MI455X (gfx1250) — matrix operand order repaired and checked by reading, not compiled, not run on hardware
//
#include <hip/hip_runtime.h>

typedef __attribute__((ext_vector_type(16))) _Float16 v16h;
typedef __attribute__((ext_vector_type(8)))  _Float16 v8h;
typedef __attribute__((ext_vector_type(8)))  float    v8f;

#define IN_F     4096
#define OUT_F    4096
#define GROUP_SZ 128
#define BM 128
#define BN 256
#define BK 64
#define LDA (BK + 8)   // 72 halves = 144B row stride (16B aligned, bank-spread)
#define LDB (BK + 8)

__global__ __launch_bounds__(256)
void gptq_wmma_kernel(const float* __restrict__ x,
                      const int*   __restrict__ qweight,
                      const int*   __restrict__ qzeros,
                      const float* __restrict__ scales,
                      const float* __restrict__ bias,
                      float*       __restrict__ out)
{
    __shared__ _Float16 Ash[BM][LDA];   // [m][k] f16
    __shared__ _Float16 Bsh[BN][LDB];   // [n][k] f16 (transposed for B fragments)

    const int tid    = threadIdx.x;
    const int lane   = tid & 31;
    const int wid    = tid >> 5;        // 0..7
    const int wave_m = wid & 1;         // 2 waves along M
    const int wave_n = wid >> 1;        // 4 waves along N
    const int l16    = lane & 15;
    const int half   = lane >> 4;       // lane group 0/1

    const int m_blk = blockIdx.y * BM;
    const int n_blk = blockIdx.x * BN;

    // accumulators: wave tile 64(M) x 64(N) = 4x4 fragments of 16x16 f32
    v8f acc[4][4];
#pragma unroll
    for (int mi = 0; mi < 4; ++mi)
#pragma unroll
        for (int ni = 0; ni < 4; ++ni)
            acc[mi][ni] = (v8f){0.f,0.f,0.f,0.f,0.f,0.f,0.f,0.f};

    // preload bias for this lane's 4 output columns
    float bi[4];
#pragma unroll
    for (int ni = 0; ni < 4; ++ni)
        bi[ni] = bias[n_blk + wave_n * 64 + ni * 16 + l16];

    // B dequant assignment: each thread owns one output column for the slab
    const int n_idx  = tid;                 // 0..255
    const int n_g    = n_blk + n_idx;       // global column
    const int zshift = (n_g & 7) * 4;

    for (int k0 = 0; k0 < IN_F; k0 += BK) {
        // ---- stage A tile: 128x64 f32 -> f16 LDS ------------------------
#pragma unroll
        for (int i = 0; i < 8; ++i) {
            const int j   = tid + i * 256;          // 0..2047 float4 chunks
            const int row = j >> 4;                 // 16 float4 per row
            const int c4  = (j & 15) << 2;
            const float4 v = *(const float4*)(x + (size_t)(m_blk + row) * IN_F + k0 + c4);
            _Float16* dst = &Ash[row][c4];
            dst[0] = (_Float16)v.x; dst[1] = (_Float16)v.y;
            dst[2] = (_Float16)v.z; dst[3] = (_Float16)v.w;
        }

        // ---- dequant B slab: 64(K) x 256(N) int4 -> f16 LDS [n][k] ------
        {
            const int   g    = k0 / GROUP_SZ;       // constant over 64-K slab
            const float s    = scales[(size_t)g * OUT_F + n_g];
            const int   qz   = qzeros[(size_t)g * (OUT_F / 8) + (n_g >> 3)];
            const int   zero = ((qz >> zshift) & 15) + 1;
            const float zs   = (float)zero * s;
            const int   krow_base = k0 >> 3;        // packed row index (8 rows)
#pragma unroll
            for (int r = 0; r < 8; ++r) {
                const unsigned w = ((const unsigned*)qweight)[(size_t)(krow_base + r) * OUT_F + n_g];
                _Float16* dst = &Bsh[n_idx][r * 8];
#pragma unroll
                for (int jj = 0; jj < 8; ++jj) {
                    const int q = (w >> (4 * jj)) & 15;
                    dst[jj] = (_Float16)((float)q * s - zs);
                }
            }
        }
        __syncthreads();

        // ---- two WMMA K-steps over the staged slab ----------------------
#pragma unroll
        for (int ks = 0; ks < 2; ++ks) {
            // A fragment (16-bit 16x32 layout): lanes 0-15: K0-7 then K16-23;
            // lanes 16-31: K8-15 then K24-31 (relative to ks*32).
            v16h afrag[4];
#pragma unroll
            for (int mi = 0; mi < 4; ++mi) {
                const _Float16* ap = &Ash[wave_m * 64 + mi * 16 + l16][ks * 32 + half * 8];
                v8h lo = *(const v8h*)(ap);
                v8h hi = *(const v8h*)(ap + 16);
                afrag[mi] = __builtin_shufflevector(lo, hi,
                    0,1,2,3,4,5,6,7,8,9,10,11,12,13,14,15);
            }
            // B fragment (32x16): lanes 0-15: K0-7 then K16-23; lanes 16-31: K8-15 then K24-31,
            // lane%16 selects column; Bsh is [n][k] so K is contiguous.
            v16h bfrag[4];
#pragma unroll
            for (int ni = 0; ni < 4; ++ni) {
                const _Float16* bp = &Bsh[wave_n * 64 + ni * 16 + l16][ks * 32 + half * 8];
                v8h lo = *(const v8h*)(bp);
                v8h hi = *(const v8h*)(bp + 16);
                bfrag[ni] = __builtin_shufflevector(lo, hi,
                    0,1,2,3,4,5,6,7,8,9,10,11,12,13,14,15);
            }

#pragma unroll
            for (int mi = 0; mi < 4; ++mi)
#pragma unroll
                for (int ni = 0; ni < 4; ++ni)
                    acc[mi][ni] = __builtin_amdgcn_wmma_f32_16x16x32_f16(
                        false, afrag[mi], false, bfrag[ni],
                        (short)0, acc[mi][ni], false, false);
        }
        __syncthreads();
    }

    // ---- epilogue: C layout VGPR v, lane l -> M = v + 8*(l>=16), N = l%16
#pragma unroll
    for (int mi = 0; mi < 4; ++mi) {
#pragma unroll
        for (int ni = 0; ni < 4; ++ni) {
            const int nc = n_blk + wave_n * 64 + ni * 16 + l16;
#pragma unroll
            for (int v = 0; v < 8; ++v) {
                const int mr = m_blk + wave_m * 64 + mi * 16 + v + half * 8;
                out[(size_t)mr * OUT_F + nc] = acc[mi][ni][v] + bi[ni];
            }
        }
    }
}

extern "C" void kernel_launch(void* const* d_in, const int* in_sizes, int n_in,
                              void* d_out, int out_size, void* d_ws, size_t ws_size,
                              hipStream_t stream) {
    const float* x       = (const float*)d_in[0];
    const int*   qweight = (const int*)  d_in[1];
    const int*   qzeros  = (const int*)  d_in[2];
    const float* scales  = (const float*)d_in[3];
    const float* bias    = (const float*)d_in[4];
    float*       out     = (float*)d_out;

    const int M = in_sizes[0] / IN_F;   // 8192
    dim3 grid(OUT_F / BN, M / BM);      // (16, 64)
    gptq_wmma_kernel<<<grid, 256, 0, stream>>>(x, qweight, qzeros, scales, bias, out);
}
